// PeepholeLSTMCell_12343736008988
// MI455X (gfx1250) — matrix operand order repaired and checked by reading, not compiled, not run on hardware
//
#include <hip/hip_runtime.h>

typedef __bf16 bf16;
typedef __attribute__((ext_vector_type(16))) __bf16 v16bf;
typedef __attribute__((ext_vector_type(8)))  __bf16 v8bf;
typedef __attribute__((ext_vector_type(8)))  float  v8f;
typedef __attribute__((ext_vector_type(4)))  int    v4i;

#define B_DIM  4096
#define IN_DIM 1024
#define H_DIM  1024
#define G_DIM  4096   // 4*H

// ---- CDNA5 feature detection (device pass only; host pass takes fallbacks) ----
#if defined(__AMDGCN__) && __has_builtin(__builtin_amdgcn_global_load_async_to_lds_b128) && \
    __has_builtin(__builtin_amdgcn_s_wait_asynccnt)
#define USE_ASYNC 1
// builtin signature (from hipcc diagnostics): (int4 AS1*, int4 AS3*, imm offset, imm cpol)
#define AS1P(p) ((__attribute__((address_space(1))) v4i*)(p))
#define AS3P(p) ((__attribute__((address_space(3))) v4i*)(p))
#else
#define USE_ASYNC 0
#endif

#if defined(__AMDGCN__) && __has_builtin(__builtin_amdgcn_sched_barrier)
#define SCHED_FENCE() __builtin_amdgcn_sched_barrier(0)
#else
#define SCHED_FENCE()
#endif

// ---------------- pre-pass 1: f32 -> bf16 (input, h) ----------------
__global__ void cvt_f32_bf16(const float* __restrict__ src, bf16* __restrict__ dst, int n8) {
    int i = blockIdx.x * blockDim.x + threadIdx.x;
    if (i >= n8) return;
    const float4* s = reinterpret_cast<const float4*>(src) + (long)i * 2;
    float4 a = s[0], b = s[1];
    v8bf o;
    o[0] = (bf16)a.x; o[1] = (bf16)a.y; o[2] = (bf16)a.z; o[3] = (bf16)a.w;
    o[4] = (bf16)b.x; o[5] = (bf16)b.y; o[6] = (bf16)b.z; o[7] = (bf16)b.w;
    reinterpret_cast<v8bf*>(dst)[i] = o;
}

// ---------------- pre-pass 2: W [4H x K] f32 -> W^T [K x 4H] bf16 ----------------
__global__ __launch_bounds__(256)
void transpose_w_bf16(const float* __restrict__ W, bf16* __restrict__ Wt) {
    __shared__ float tile[32][33];
    const int tx = threadIdx.x & 31, ty = threadIdx.x >> 5;   // 32 x 8
    const int bk = blockIdx.x * 32;   // K base   (cols of W)
    const int bj = blockIdx.y * 32;   // row base (rows of W, i.e. 4H dim)
#pragma unroll
    for (int r = 0; r < 32; r += 8)
        tile[ty + r][tx] = W[(long)(bj + ty + r) * IN_DIM + bk + tx];
    __syncthreads();
#pragma unroll
    for (int r = 0; r < 32; r += 8)
        Wt[(long)(bk + ty + r) * G_DIM + bj + tx] = (bf16)tile[tx][ty + r];
}

// ---------------- fused peephole-LSTM GEMM + epilogue ----------------
#define BM     128   // batch rows per block
#define BN     32    // hidden cols per block (x4 gates internally)
#define KT     32    // K per step (bf16 WMMA K)
#define APITCH 40    // lds A row pitch, elems (32 + 8 pad -> 80B, 16B aligned)
#define WPITCH 136   // lds Wt row pitch, elems (128 + 8 pad -> 272B, 16B aligned)
#define NSTEPS 64    // 32 steps over input/W_ih^T + 32 steps over h/W_hh^T

union FragBf { v16bf v; uint4 q[2]; };

__global__ __launch_bounds__(256)
void lstm_wmma_kernel(const bf16* __restrict__ Xbf, const bf16* __restrict__ Hbf,
                      const bf16* __restrict__ WihT, const bf16* __restrict__ WhhT,
                      const float* __restrict__ c_in,
                      const float* __restrict__ bias_ih, const float* __restrict__ bias_hh,
                      const float* __restrict__ wic, const float* __restrict__ wfc,
                      const float* __restrict__ woc,
                      float* __restrict__ out /* [h_new | c_new] */) {
    __shared__ bf16 lds_a[2][BM * APITCH];   // A tile: [row][k]
    __shared__ bf16 lds_w[2][KT * WPITCH];   // Wt tile: [k][jj], jj = gate*32 + local_j

    const int tid  = threadIdx.x;
    const int lane = tid & 31;
    const int wave = tid >> 5;
    const int wm0  = (wave >> 1) * 32;   // 4 waves in M
    const int wn0  = (wave & 1) * 16;    // 2 waves in N
    const int bm0  = blockIdx.x * BM;
    const int bn0  = blockIdx.y * BN;

    // ---- staging assignment: two 16B chunks per thread per tile
    // A tile 128x32: chunk c -> row c>>2, k-elem (c&3)*8
    const int ar0 = tid >> 2, ak0 = (tid & 3) * 8;
    const int ar1 = ar0 + 64;
    // Wt tile 32x(4x32): chunk c -> k row c>>4, sub c&15 -> gate sub>>2, part sub&3
    const int wk0 = tid >> 4, wsub = tid & 15;
    const int wk1 = wk0 + 16;
    const int wgcolG = (wsub >> 2) * H_DIM + bn0 + (wsub & 3) * 8;  // global col in [4H]
    const int wgcolL = (wsub >> 2) * 32 + (wsub & 3) * 8;           // lds col  in [128]

    const long aoffg0 = (long)(bm0 + ar0) * IN_DIM + ak0;
    const long aoffg1 = (long)(bm0 + ar1) * IN_DIM + ak0;

    v8f acc[2][4];
    const v8f vzero = {};
#pragma unroll
    for (int mt = 0; mt < 2; ++mt)
#pragma unroll
        for (int g = 0; g < 4; ++g) acc[mt][g] = vzero;

    auto stage = [&](int s, int buf) {
        const bf16* A  = (s < 32) ? Xbf  : Hbf;
        const bf16* Wt = (s < 32) ? WihT : WhhT;
        const int k0 = (s & 31) * KT;
        const bf16* ga0 = A + aoffg0 + k0;
        const bf16* ga1 = A + aoffg1 + k0;
        const bf16* gw0 = Wt + (long)(k0 + wk0) * G_DIM + wgcolG;
        const bf16* gw1 = Wt + (long)(k0 + wk1) * G_DIM + wgcolG;
        bf16* la0 = &lds_a[buf][ar0 * APITCH + ak0];
        bf16* la1 = &lds_a[buf][ar1 * APITCH + ak0];
        bf16* lw0 = &lds_w[buf][wk0 * WPITCH + wgcolL];
        bf16* lw1 = &lds_w[buf][wk1 * WPITCH + wgcolL];
#if USE_ASYNC
        __builtin_amdgcn_global_load_async_to_lds_b128(AS1P(ga0), AS3P(la0), 0, 0);
        __builtin_amdgcn_global_load_async_to_lds_b128(AS1P(ga1), AS3P(la1), 0, 0);
        __builtin_amdgcn_global_load_async_to_lds_b128(AS1P(gw0), AS3P(lw0), 0, 0);
        __builtin_amdgcn_global_load_async_to_lds_b128(AS1P(gw1), AS3P(lw1), 0, 0);
#else
        *reinterpret_cast<uint4*>(la0) = *reinterpret_cast<const uint4*>(ga0);
        *reinterpret_cast<uint4*>(la1) = *reinterpret_cast<const uint4*>(ga1);
        *reinterpret_cast<uint4*>(lw0) = *reinterpret_cast<const uint4*>(gw0);
        *reinterpret_cast<uint4*>(lw1) = *reinterpret_cast<const uint4*>(gw1);
#endif
    };

    stage(0, 0);
#if USE_ASYNC
    __builtin_amdgcn_s_wait_asynccnt(0);
#endif
    __syncthreads();

    // ---- per-lane fragment LDS element offsets (ISA 16-bit A / B layouts, wave32)
    const int arow = lane & 15;
    const int asel = (lane >> 4) * 8;  // lanes 0-15: K {0..7,16..23}; lanes 16-31: {8..15,24..31}
    const int aoff0 = (wm0 +      arow) * APITCH + asel;
    const int aoff1 = (wm0 + 16 + arow) * APITCH + asel;
    const int boffB = wn0 + (lane & 15);  // B: lane = N col, K in same order as A

    // unroll 1: keep loop-carried accumulators pinned in one register set
    // (unrolling makes regalloc rotate the tied C/D operands -> v_mov storms)
#pragma unroll 1
    for (int s = 0; s < NSTEPS; ++s) {
        if (s + 1 < NSTEPS) stage(s + 1, (s + 1) & 1);

        const bf16* ba = &lds_a[s & 1][0];
        const bf16* bw = &lds_w[s & 1][0];
        FragBf a0, a1, bfr[4];
        a0.q[0] = *reinterpret_cast<const uint4*>(&ba[aoff0]);
        a0.q[1] = *reinterpret_cast<const uint4*>(&ba[aoff0 + 16]);
        a1.q[0] = *reinterpret_cast<const uint4*>(&ba[aoff1]);
        a1.q[1] = *reinterpret_cast<const uint4*>(&ba[aoff1 + 16]);
#pragma unroll
        for (int g = 0; g < 4; ++g) {
#pragma unroll
            for (int e = 0; e < 16; ++e)
                bfr[g].v[e] = bw[boffB + g * 32 + (asel + (e & 7) + (e >> 3) * 16) * WPITCH];
        }
        SCHED_FENCE();   // keep all 12 fragment loads issued before the WMMA block
#pragma unroll
        for (int g = 0; g < 4; ++g) {
            acc[0][g] = __builtin_amdgcn_wmma_f32_16x16x32_bf16(
                false, a0.v, false, bfr[g].v, (short)0, acc[0][g], false, false);
            acc[1][g] = __builtin_amdgcn_wmma_f32_16x16x32_bf16(
                false, a1.v, false, bfr[g].v, (short)0, acc[1][g], false, false);
        }
#if USE_ASYNC
        __builtin_amdgcn_s_wait_asynccnt(0);
#endif
        __syncthreads();
    }

    // ---- fused epilogue: peephole gates + state update
    // C/D layout: VGPR r, lane l -> M = r + 8*(l>=16), N = l&15
    const int j     = bn0 + wn0 + (lane & 15);
    const int mbase = bm0 + wm0 + ((lane >> 4) ? 8 : 0);

    float bsum[4];
#pragma unroll
    for (int g = 0; g < 4; ++g) bsum[g] = bias_ih[g * H_DIM + j] + bias_hh[g * H_DIM + j];
    const float pwi = wic[j], pwf = wfc[j], pwo = woc[j];

    float* out_h = out;
    float* out_c = out + (long)B_DIM * H_DIM;

#pragma unroll
    for (int mt = 0; mt < 2; ++mt) {
#pragma unroll
        for (int r = 0; r < 8; ++r) {
            const long row = mbase + mt * 16 + r;
            const float cv = c_in[row * H_DIM + j];
            const float xi = acc[mt][0][r] + bsum[0] + cv * pwi;
            const float xf = acc[mt][1][r] + bsum[1] + cv * pwf;
            const float xg = acc[mt][2][r] + bsum[2];
            const float xo = acc[mt][3][r] + bsum[3] + cv * pwo;
            const float ig = 1.0f / (1.0f + __expf(-xi));
            const float fg = 1.0f / (1.0f + __expf(-xf));
            const float gg = tanhf(xg);
            const float og = 1.0f / (1.0f + __expf(-xo));
            const float cn = fg * cv + ig * gg;
            const float hn = og * tanhf(cn);
            out_h[row * H_DIM + j] = hn;
            out_c[row * H_DIM + j] = cn;
        }
    }
}

// ---------------- host launcher ----------------
extern "C" void kernel_launch(void* const* d_in, const int* in_sizes, int n_in,
                              void* d_out, int out_size, void* d_ws, size_t ws_size,
                              hipStream_t stream) {
    const float* input = (const float*)d_in[0];
    const float* h     = (const float*)d_in[1];
    const float* c     = (const float*)d_in[2];
    const float* wih   = (const float*)d_in[3];
    const float* whh   = (const float*)d_in[4];
    const float* bih   = (const float*)d_in[5];
    const float* bhh   = (const float*)d_in[6];
    const float* wic   = (const float*)d_in[7];
    const float* wfc   = (const float*)d_in[8];
    const float* woc   = (const float*)d_in[9];
    float* out = (float*)d_out;

    // workspace: x_bf | h_bf | WihT | WhhT  (4 x 4M bf16 = 32 MB)
    const long N4M = (long)4 * 1024 * 1024;
    bf16* x_bf   = (bf16*)d_ws;
    bf16* h_bf   = x_bf + N4M;
    bf16* wihT   = x_bf + 2 * N4M;
    bf16* whhT   = x_bf + 3 * N4M;

    const int n8 = (int)(N4M / 8);
    cvt_f32_bf16<<<dim3(n8 / 256), dim3(256), 0, stream>>>(input, x_bf, n8);
    cvt_f32_bf16<<<dim3(n8 / 256), dim3(256), 0, stream>>>(h,     h_bf, n8);

    dim3 tgrid(IN_DIM / 32, G_DIM / 32);
    transpose_w_bf16<<<tgrid, dim3(256), 0, stream>>>(wih, wihT);
    transpose_w_bf16<<<tgrid, dim3(256), 0, stream>>>(whh, whhT);

    dim3 grid(B_DIM / BM, H_DIM / BN), block(256);
    lstm_wmma_kernel<<<grid, block, 0, stream>>>(x_bf, h_bf, wihT, whhT,
                                                 c, bih, bhh, wic, wfc, woc, out);
}
